// MyViT_19198503813666
// MI455X (gfx1250) — matrix operand order repaired and checked by reading, not compiled, not run on hardware
//
#include <hip/hip_runtime.h>
#include <hip/hip_bf16.h>

// ---------------------------------------------------------------------------
// ViT forward for MI455X (gfx1250, wave32). All matmuls via
// v_wmma_f32_16x16x32_bf16; softmax/LN/GELU in fp32 VALU.
// B=8, S=1024, P=128, D=256, H=4, DH=64, L=4, DF=1024, O=10.
// Sequence padded 1025 -> SP=1056 (66*16 = 33*32) with key masking.
// ---------------------------------------------------------------------------

typedef __bf16 bf16_t;
typedef bf16_t v16bf __attribute__((ext_vector_type(16)));
typedef bf16_t v8bf  __attribute__((ext_vector_type(8)));
typedef float  v8f   __attribute__((ext_vector_type(8)));

#define NB     8
#define SEQ    1024
#define NPATCH 128
#define DMODEL 256
#define NH     4
#define DHEAD  64
#define NLAYER 4
#define DFF    1024
#define NOUT   10
#define NTOK   1025          // 1 cls + 1024 patches
#define SP     1056          // padded rows (66*16, 33*32)

enum { MODE_F32_BIAS = 0, MODE_ADD_F32 = 1, MODE_BF16 = 2, MODE_BF16_T = 3, MODE_BF16_GELU = 4 };

__device__ __forceinline__ bf16_t f2bf(float f) {
  unsigned u = __float_as_uint(f);
  u += 0x7FFFu + ((u >> 16) & 1u);           // round-to-nearest-even
  unsigned short h = (unsigned short)(u >> 16);
  return __builtin_bit_cast(bf16_t, h);
}

__device__ __forceinline__ v16bf cat8(v8bf lo, v8bf hi) {
  return __builtin_shufflevector(lo, hi, 0, 1, 2, 3, 4, 5, 6, 7, 8, 9, 10, 11, 12, 13, 14, 15);
}

__device__ __forceinline__ v8f wmma_bf16(v16bf a, v16bf b, v8f c) {
  return __builtin_amdgcn_wmma_f32_16x16x32_bf16(false, a, false, b, (short)0, c, false, false);
}

// ---------------------------------------------------------------------------
// fp32 -> bf16 conversion
// ---------------------------------------------------------------------------
__global__ void cvt_f32_bf16_kernel(const float* __restrict__ s, bf16_t* __restrict__ d, int n) {
  int i = blockIdx.x * blockDim.x + threadIdx.x;
  if (i < n) d[i] = f2bf(s[i]);
}

// ---------------------------------------------------------------------------
// Generic bf16 GEMM: C[M,N] = A[M,K] @ Bt[N,K]^T (+bias) with epilogue modes.
// One wave computes a 16x32 output tile (A frag reused across 2 B frags).
// Batched over gridDim.y (b) and gridDim.z (h) with separate strides.
// A-frag:  lane m=L&15, half=L>>4 : k in [k0+half*16, +16), row m of A
// B-frag:  lane n=L&15, half     : k in [k0+half*16, +16), row n of Bt
// C-frag:  vgpr r -> row half*8+r, col n=L&15
// ---------------------------------------------------------------------------
__global__ __launch_bounds__(128) void gemm_bf16_kernel(
    const bf16_t* __restrict__ A, long long sAb, long long sAh, int lda,
    const bf16_t* __restrict__ Bt, long long sBh, int ldb,
    const float* __restrict__ bias, int sBiasH,
    float* __restrict__ Cf, bf16_t* __restrict__ Cb,
    long long sCb, long long sCh, int ldc,
    int Mt, int Nt2, int K, int mode, float scale)
{
  int wave = threadIdx.x >> 5;
  int lane = threadIdx.x & 31;
  int tile = blockIdx.x * 4 + wave;
  if (tile >= Mt * Nt2) return;
  int tm  = tile / Nt2;
  int tn2 = tile % Nt2;

  int m_ = lane & 15, half = lane >> 4;

  const bf16_t* Ab = A + (long long)blockIdx.y * sAb + (long long)blockIdx.z * sAh
                       + (long long)(tm * 16 + m_) * lda + half * 16;
  const bf16_t* B0 = Bt + (long long)blockIdx.z * sBh
                        + (long long)(tn2 * 32 + m_) * ldb + half * 16;
  const bf16_t* B1 = B0 + (long long)16 * ldb;

  v8f c0 = {}, c1 = {};
  for (int k0 = 0; k0 < K; k0 += 32) {
    v8bf alo = *(const v8bf*)(Ab + k0);
    v8bf ahi = *(const v8bf*)(Ab + k0 + 8);
    v16bf a  = cat8(alo, ahi);
    v16bf b0 = *(const v16bf*)(B0 + k0);
    v16bf b1 = *(const v16bf*)(B1 + k0);
    c0 = wmma_bf16(a, b0, c0);
    c1 = wmma_bf16(a, b1, c1);
  }

  long long cBase = (long long)blockIdx.y * sCb + (long long)blockIdx.z * sCh;
  int rowBase = tm * 16 + (half ? 8 : 0);
  const float* bp_ = bias ? (bias + (long long)blockIdx.z * sBiasH) : nullptr;

#pragma unroll
  for (int t = 0; t < 2; ++t) {
    v8f cc = t ? c1 : c0;
    int nglob = tn2 * 32 + t * 16 + m_;
    float bv = bp_ ? bp_[nglob] : 0.f;
#pragma unroll
    for (int r = 0; r < 8; ++r) {
      int mg = rowBase + r;
      float v = (cc[r] + bv) * scale;
      long long ci = cBase + (long long)mg * ldc + nglob;
      switch (mode) {
        case MODE_F32_BIAS: Cf[ci] = v; break;
        case MODE_ADD_F32:  Cf[ci] += v; break;
        case MODE_BF16:     Cb[ci] = f2bf(v); break;
        case MODE_BF16_T:   Cb[cBase + (long long)nglob * ldc + mg] = f2bf(v); break;
        case MODE_BF16_GELU: {
          float g = 0.5f * v * (1.0f + erff(v * 0.70710678118654752f));
          Cb[ci] = f2bf(g);
        } break;
      }
    }
  }
}

// ---------------------------------------------------------------------------
// Embed fixup: row0 = cls + pos[0]; rows 1..1024 += pos[row]; pad rows = 0.
// ---------------------------------------------------------------------------
__global__ void embed_fix_kernel(float* __restrict__ X, const float* __restrict__ cls,
                                 const float* __restrict__ pos)
{
  long long i = (long long)blockIdx.x * blockDim.x + threadIdx.x;
  if (i >= (long long)NB * SP * DMODEL) return;
  int d = (int)(i % DMODEL);
  int r = (int)((i / DMODEL) % SP);
  if (r == 0)          X[i] = cls[d] + pos[d];
  else if (r < NTOK)   X[i] += pos[(long long)r * DMODEL + d];
  else                 X[i] = 0.f;
}

// ---------------------------------------------------------------------------
// LayerNorm over D=256, one wave per row, output bf16.
// ---------------------------------------------------------------------------
__global__ __launch_bounds__(128) void ln_kernel(
    const float* __restrict__ X, bf16_t* __restrict__ Hb,
    const float* __restrict__ g, const float* __restrict__ b, int nrows)
{
  int wave = threadIdx.x >> 5, lane = threadIdx.x & 31;
  int row = blockIdx.x * 4 + wave;
  if (row >= nrows) return;
  const float* xr = X + (long long)row * DMODEL;
  float v[8];
#pragma unroll
  for (int j = 0; j < 8; ++j) v[j] = xr[lane + 32 * j];
  float s = 0.f;
#pragma unroll
  for (int j = 0; j < 8; ++j) s += v[j];
#pragma unroll
  for (int off = 16; off > 0; off >>= 1) s += __shfl_xor(s, off, 32);
  float mean = s * (1.f / DMODEL);
  float q = 0.f;
#pragma unroll
  for (int j = 0; j < 8; ++j) { v[j] -= mean; q += v[j] * v[j]; }
#pragma unroll
  for (int off = 16; off > 0; off >>= 1) q += __shfl_xor(q, off, 32);
  float inv = rsqrtf(q * (1.f / DMODEL) + 1e-5f);
  bf16_t* hr = Hb + (long long)row * DMODEL;
#pragma unroll
  for (int j = 0; j < 8; ++j) {
    int c = lane + 32 * j;
    hr[c] = f2bf(v[j] * inv * g[c] + b[c]);
  }
}

// ---------------------------------------------------------------------------
// Flash attention: one wave per (b, h, 16-query tile). Online softmax over
// 33 key-steps of 32 keys. Q pre-scaled by 1/8. Adds result into X (residual).
// Q,K: [B,H,SP,64] bf16 ; Vt: [B,H,64,SP] bf16.
// ---------------------------------------------------------------------------
__global__ __launch_bounds__(32) void attn_kernel(
    const bf16_t* __restrict__ Q, const bf16_t* __restrict__ Kb,
    const bf16_t* __restrict__ Vt, float* __restrict__ X)
{
  __shared__ bf16_t Pt[16 * 32];
  int qt = blockIdx.x, h = blockIdx.y, b = blockIdx.z;
  int lane = threadIdx.x;
  int m_ = lane & 15, half = lane >> 4;
  long long headOff = ((long long)b * NH + h) * (long long)SP * DHEAD;

  const bf16_t* qp = Q + headOff + (long long)(qt * 16 + m_) * DHEAD + half * 16;
  const bf16_t* kp = Kb + headOff;
  const bf16_t* vp = Vt + headOff;   // [DHEAD, SP]

  v16bf qa0 = cat8(*(const v8bf*)(qp),      *(const v8bf*)(qp + 8));   // k 0..31
  v16bf qa1 = cat8(*(const v8bf*)(qp + 32), *(const v8bf*)(qp + 40)); // k 32..63

  v8f o0 = {}, o1 = {}, o2 = {}, o3 = {};
  float mrun[8], lrun[8];
#pragma unroll
  for (int r = 0; r < 8; ++r) { mrun[r] = -1e30f; lrun[r] = 0.f; }

  for (int kt = 0; kt < SP / 32; ++kt) {
    int key0 = kt * 32;
    // ---- scores: two 16x16 tiles (keys key0..+15, key0+16..+31) ----
    const bf16_t* kr0 = kp + (long long)(key0 + m_) * DHEAD + half * 16;
    const bf16_t* kr1 = kr0 + 16 * DHEAD;
    v8f c0 = {}, c1 = {};
    c0 = wmma_bf16(qa0, *(const v16bf*)(kr0),      c0);
    c0 = wmma_bf16(qa1, *(const v16bf*)(kr0 + 32), c0);
    c1 = wmma_bf16(qa0, *(const v16bf*)(kr1),      c1);
    c1 = wmma_bf16(qa1, *(const v16bf*)(kr1 + 32), c1);

    // ---- masked online softmax ----
    bool ok0 = (key0 + m_) < NTOK;
    bool ok1 = (key0 + 16 + m_) < NTOK;
    float p0[8], p1[8], rm[8], rs[8];
#pragma unroll
    for (int r = 0; r < 8; ++r) {
      p0[r] = ok0 ? c0[r] : -1e30f;
      p1[r] = ok1 ? c1[r] : -1e30f;
      rm[r] = fmaxf(p0[r], p1[r]);
    }
#pragma unroll
    for (int off = 1; off < 16; off <<= 1) {
#pragma unroll
      for (int r = 0; r < 8; ++r) rm[r] = fmaxf(rm[r], __shfl_xor(rm[r], off, 32));
    }
#pragma unroll
    for (int r = 0; r < 8; ++r) {
      float mnew = fmaxf(mrun[r], rm[r]);
      float sc = __expf(mrun[r] - mnew);
      mrun[r] = mnew;
      p0[r] = __expf(p0[r] - mnew);
      p1[r] = __expf(p1[r] - mnew);
      rs[r] = p0[r] + p1[r];
      lrun[r] *= sc;
      o0[r] *= sc; o1[r] *= sc; o2[r] *= sc; o3[r] *= sc;
    }
#pragma unroll
    for (int off = 1; off < 16; off <<= 1) {
#pragma unroll
      for (int r = 0; r < 8; ++r) rs[r] += __shfl_xor(rs[r], off, 32);
    }
#pragma unroll
    for (int r = 0; r < 8; ++r) lrun[r] += rs[r];

    // ---- C-layout -> A-layout via LDS bounce (same wave, DS in-order) ----
    int prow = half ? 8 : 0;
#pragma unroll
    for (int r = 0; r < 8; ++r) {
      Pt[(prow + r) * 32 + m_]      = f2bf(p0[r]);
      Pt[(prow + r) * 32 + 16 + m_] = f2bf(p1[r]);
    }
    v16bf pa = cat8(*(const v8bf*)(&Pt[m_ * 32 + half * 16]),
                    *(const v8bf*)(&Pt[m_ * 32 + half * 16 + 8]));

    // ---- o += P @ V  (V^T rows = head dims, contiguous keys) ----
    const bf16_t* vr = vp + (long long)m_ * SP + key0 + half * 16;
    o0 = wmma_bf16(pa, *(const v16bf*)(vr),               o0);
    o1 = wmma_bf16(pa, *(const v16bf*)(vr + 16 * SP),     o1);
    o2 = wmma_bf16(pa, *(const v16bf*)(vr + 32 * SP),     o2);
    o3 = wmma_bf16(pa, *(const v16bf*)(vr + 48 * SP),     o3);
  }

  // ---- normalize and residual-add into X ----
#pragma unroll
  for (int r = 0; r < 8; ++r) {
    int srow = qt * 16 + (half ? 8 : 0) + r;
    if (srow < NTOK) {
      float inv = 1.f / lrun[r];
      long long xb = ((long long)b * SP + srow) * DMODEL + h * DHEAD + m_;
      X[xb]      += o0[r] * inv;
      X[xb + 16] += o1[r] * inv;
      X[xb + 32] += o2[r] * inv;
      X[xb + 48] += o3[r] * inv;
    }
  }
}

// ---------------------------------------------------------------------------
// Head: logits = x[:,0] @ Wo^T + bo, softmax. One wave per batch row.
// ---------------------------------------------------------------------------
__global__ __launch_bounds__(32) void head_kernel(
    const float* __restrict__ X, const float* __restrict__ Wo,
    const float* __restrict__ bo, float* __restrict__ out)
{
  int b = blockIdx.x, lane = threadIdx.x;
  __shared__ float e[16];
  if (lane < NOUT) {
    float acc = bo[lane];
    const float* xr = X + (long long)b * SP * DMODEL;
    const float* wr = Wo + lane * DMODEL;
    for (int d = 0; d < DMODEL; ++d) acc += xr[d] * wr[d];
    e[lane] = acc;
  }
  __syncthreads();
  if (lane == 0) {
    float mx = e[0];
    for (int i = 1; i < NOUT; ++i) mx = fmaxf(mx, e[i]);
    float s = 0.f, ex[NOUT];
    for (int i = 0; i < NOUT; ++i) { ex[i] = __expf(e[i] - mx); s += ex[i]; }
    float inv = 1.f / s;
    for (int i = 0; i < NOUT; ++i) out[b * NOUT + i] = ex[i] * inv;
  }
}

// ---------------------------------------------------------------------------
// Host orchestration
// ---------------------------------------------------------------------------
extern "C" void kernel_launch(void* const* d_in, const int* in_sizes, int n_in,
                              void* d_out, int out_size, void* d_ws, size_t ws_size,
                              hipStream_t stream)
{
  (void)in_sizes; (void)n_in; (void)out_size; (void)ws_size;
  const float* images = (const float*)d_in[0];
  const float* Wp  = (const float*)d_in[1];
  const float* bp  = (const float*)d_in[2];
  const float* cls = (const float*)d_in[3];
  const float* pos = (const float*)d_in[4];
  const float* ln1_g = (const float*)d_in[5];
  const float* ln1_b = (const float*)d_in[6];
  const float* Wq = (const float*)d_in[7];
  const float* bq = (const float*)d_in[8];
  const float* Wk = (const float*)d_in[9];
  const float* bk = (const float*)d_in[10];
  const float* Wv = (const float*)d_in[11];
  const float* bv = (const float*)d_in[12];
  const float* ln2_g = (const float*)d_in[13];
  const float* ln2_b = (const float*)d_in[14];
  const float* W1 = (const float*)d_in[15];
  const float* b1 = (const float*)d_in[16];
  const float* W2 = (const float*)d_in[17];
  const float* b2 = (const float*)d_in[18];
  const float* Wo = (const float*)d_in[19];
  const float* bo = (const float*)d_in[20];
  float* out = (float*)d_out;

  char* ws = (char*)d_ws;
  size_t off = 0;
  auto alloc = [&](size_t bytes) -> char* {
    char* p = ws + off; off += (bytes + 255) & ~(size_t)255; return p;
  };
  float*  X    = (float*) alloc((size_t)NB * SP * DMODEL * 4);
  bf16_t* Hb   = (bf16_t*)alloc((size_t)NB * SP * DMODEL * 2);
  bf16_t* Qb   = (bf16_t*)alloc((size_t)NB * NH * SP * DHEAD * 2);
  bf16_t* Kbf  = (bf16_t*)alloc((size_t)NB * NH * SP * DHEAD * 2);
  bf16_t* Vt   = (bf16_t*)alloc((size_t)NB * NH * SP * DHEAD * 2);
  bf16_t* Hmid = (bf16_t*)alloc((size_t)NB * SP * DFF * 2);
  bf16_t* IMGb = (bf16_t*)alloc((size_t)NB * SEQ * NPATCH * 2);
  bf16_t* Wpb  = (bf16_t*)alloc((size_t)DMODEL * NPATCH * 2);
  bf16_t* Wqb  = (bf16_t*)alloc((size_t)NLAYER * NH * DHEAD * DHEAD * 2);
  bf16_t* Wkb  = (bf16_t*)alloc((size_t)NLAYER * NH * DHEAD * DHEAD * 2);
  bf16_t* Wvb  = (bf16_t*)alloc((size_t)NLAYER * NH * DHEAD * DHEAD * 2);
  bf16_t* W1b  = (bf16_t*)alloc((size_t)NLAYER * DFF * DMODEL * 2);
  bf16_t* W2b  = (bf16_t*)alloc((size_t)NLAYER * DMODEL * DFF * 2);

  auto cvt = [&](const float* s, bf16_t* d, int n) {
    cvt_f32_bf16_kernel<<<(n + 255) / 256, 256, 0, stream>>>(s, d, n);
  };
  cvt(images, IMGb, NB * SEQ * NPATCH);
  cvt(Wp, Wpb, DMODEL * NPATCH);
  cvt(Wq, Wqb, NLAYER * NH * DHEAD * DHEAD);
  cvt(Wk, Wkb, NLAYER * NH * DHEAD * DHEAD);
  cvt(Wv, Wvb, NLAYER * NH * DHEAD * DHEAD);
  cvt(W1, W1b, NLAYER * DFF * DMODEL);
  cvt(W2, W2b, NLAYER * DMODEL * DFF);

  // Patch embed: tokens[b, 1024, 256] = images @ Wp^T + bp, into X rows 1..1024
  gemm_bf16_kernel<<<dim3((64 * 8) / 4, NB, 1), 128, 0, stream>>>(
      IMGb, (long long)SEQ * NPATCH, 0, NPATCH,
      Wpb, 0, NPATCH,
      bp, 0,
      X + DMODEL, nullptr,
      (long long)SP * DMODEL, 0, DMODEL,
      64, 8, NPATCH, MODE_F32_BIAS, 1.0f);

  {
    long long total = (long long)NB * SP * DMODEL;
    embed_fix_kernel<<<(unsigned)((total + 255) / 256), 256, 0, stream>>>(X, cls, pos);
  }

  const int nrows = NB * SP;                 // 8448
  for (int l = 0; l < NLAYER; ++l) {
    // LN1 -> Hb (bf16)
    ln_kernel<<<nrows / 4, 128, 0, stream>>>(X, Hb, ln1_g + l * DMODEL, ln1_b + l * DMODEL, nrows);

    // Q/K/V per (b,h): [SP,64] = Hb[:, h*64:+64] @ W^T + bias
    long long sWl  = (long long)NH * DHEAD * DHEAD;
    long long sBl  = (long long)NH * DHEAD;
    dim3 qkvGrid((66 * 2) / 4, NB, NH);
    gemm_bf16_kernel<<<qkvGrid, 128, 0, stream>>>(
        Hb, (long long)SP * DMODEL, DHEAD, DMODEL,
        Wqb + l * sWl, (long long)DHEAD * DHEAD, DHEAD,
        bq + l * sBl, DHEAD,
        nullptr, Qb, (long long)NH * SP * DHEAD, (long long)SP * DHEAD, DHEAD,
        66, 2, DHEAD, MODE_BF16, 0.125f);           // fold 1/sqrt(64)
    gemm_bf16_kernel<<<qkvGrid, 128, 0, stream>>>(
        Hb, (long long)SP * DMODEL, DHEAD, DMODEL,
        Wkb + l * sWl, (long long)DHEAD * DHEAD, DHEAD,
        bk + l * sBl, DHEAD,
        nullptr, Kbf, (long long)NH * SP * DHEAD, (long long)SP * DHEAD, DHEAD,
        66, 2, DHEAD, MODE_BF16, 1.0f);
    gemm_bf16_kernel<<<qkvGrid, 128, 0, stream>>>(
        Hb, (long long)SP * DMODEL, DHEAD, DMODEL,
        Wvb + l * sWl, (long long)DHEAD * DHEAD, DHEAD,
        bv + l * sBl, DHEAD,
        nullptr, Vt, (long long)NH * DHEAD * SP, (long long)DHEAD * SP, SP,
        66, 2, DHEAD, MODE_BF16_T, 1.0f);           // store V transposed

    // Attention + residual into X
    attn_kernel<<<dim3(SP / 16, NH, NB), 32, 0, stream>>>(Qb, Kbf, Vt, X);

    // LN2 -> Hb
    ln_kernel<<<nrows / 4, 128, 0, stream>>>(X, Hb, ln2_g + l * DMODEL, ln2_b + l * DMODEL, nrows);

    // MLP1: Hmid = gelu(Hb @ W1^T + b1)   [8448,1024]
    gemm_bf16_kernel<<<dim3((528 * 32) / 4, 1, 1), 128, 0, stream>>>(
        Hb, 0, 0, DMODEL,
        W1b + (long long)l * DFF * DMODEL, 0, DMODEL,
        b1 + l * DFF, 0,
        nullptr, Hmid, 0, 0, DFF,
        528, 32, DMODEL, MODE_BF16_GELU, 1.0f);

    // MLP2: X += Hmid @ W2^T + b2          [8448,256]
    gemm_bf16_kernel<<<dim3((528 * 8) / 4, 1, 1), 128, 0, stream>>>(
        Hmid, 0, 0, DFF,
        W2b + (long long)l * DMODEL * DFF, 0, DFF,
        b2 + l * DMODEL, 0,
        X, nullptr, 0, 0, DMODEL,
        528, 8, DFF, MODE_ADD_F32, 1.0f);
  }

  head_kernel<<<NB, 32, 0, stream>>>(X, Wo, bo, out);
}
